// MultiHeadAttention_49065706389646
// MI455X (gfx1250) — matrix operand order repaired and checked by reading, not compiled, not run on hardware
//
#include <hip/hip_runtime.h>
#include <math.h>

typedef __attribute__((ext_vector_type(16))) _Float16 v16h;
typedef __attribute__((ext_vector_type(8)))  float    v8f;

// Problem constants
static constexpr int NB = 8;     // batch
static constexpr int SS = 2048;  // sequence
static constexpr int DD = 768;   // model dim

// padded LDS row stride in halfs: 80B -> bank stride 20, conflict-free frag loads
static constexpr int LROW = 40;

#if __has_builtin(__builtin_amdgcn_global_load_async_to_lds_b128) && \
    __has_builtin(__builtin_amdgcn_s_wait_asynccnt)
#define USE_ASYNC_LDS 1
typedef int vint4 __attribute__((ext_vector_type(4)));
typedef __attribute__((address_space(1))) vint4* as1_v4p;
typedef __attribute__((address_space(3))) vint4* as3_v4p;
#else
#define USE_ASYNC_LDS 0
#endif

// ---------------------------------------------------------------------------
// helpers
// ---------------------------------------------------------------------------
__device__ inline v16h frag16(uint4 lo, uint4 hi) {
    union { uint4 u[2]; v16h v; } t;
    t.u[0] = lo; t.u[1] = hi;
    return t.v;
}

__device__ inline v8f wmma16(v16h a, v16h b, v8f c) {
    // D = A(16x32 f16) * B(32x16 f16) + C(16x16 f32)
    return __builtin_amdgcn_wmma_f32_16x16x32_f16(
        /*neg_a=*/false, a, /*neg_b=*/false, b,
        /*c_mod=*/(short)0, c, /*reuse_a=*/false, /*reuse_b=*/false);
}

// ---------------------------------------------------------------------------
// f32 -> f16 conversion (vectorized x4)
// ---------------------------------------------------------------------------
__global__ __launch_bounds__(256)
void cvt_f32_to_f16(const float* __restrict__ in, _Float16* __restrict__ out, int n)
{
    int i = (blockIdx.x * 256 + threadIdx.x) * 4;
    if (i < n) {
        float4 v = *(const float4*)(in + i);
        union { _Float16 h[4]; uint2 u; } pk;
        pk.h[0] = (_Float16)v.x; pk.h[1] = (_Float16)v.y;
        pk.h[2] = (_Float16)v.z; pk.h[3] = (_Float16)v.w;
        *(uint2*)(out + i) = pk.u;
    }
}

// ---------------------------------------------------------------------------
// W[e][d] f32  ->  Wt[d][e] f16   (tiled 32x32 transpose)
// ---------------------------------------------------------------------------
__global__ __launch_bounds__(256)
void transpose_to_f16(const float* __restrict__ W, _Float16* __restrict__ Wt, int Dn)
{
    __shared__ float t[32][33];
    int tx = threadIdx.x & 31;
    int ty = threadIdx.x >> 5;            // 0..7
    int e0 = blockIdx.x * 32, d0 = blockIdx.y * 32;
    #pragma unroll
    for (int r = 0; r < 32; r += 8)
        t[ty + r][tx] = W[(size_t)(e0 + ty + r) * Dn + d0 + tx];
    __syncthreads();
    #pragma unroll
    for (int r = 0; r < 32; r += 8)
        Wt[(size_t)(d0 + ty + r) * Dn + e0 + tx] = (_Float16)t[tx][ty + r];
}

// ---------------------------------------------------------------------------
// Generic WMMA GEMM:  C[M,N] = scale * (A[M,K] @ B[K,N]) + bias
//   A: f16 row-major (lda), B: f16 row-major (ldb)
//   mode 0: C f32 row-major (ldc), += bias[n] if bias
//   mode 1: C f16 row-major (ldc), += bias[n] if bias
//   mode 2: C f16 transposed, batched: Ct[b][n][m%ldc], ldc = S per batch,
//           batch stride N*ldc (used for K^T).  += bias[n]
// 256 threads = 8 waves; workgroup tile 128(M) x 128(N); BK = 32.
// Wave tile 32x64 (2x4 accumulators). Double-buffered LDS, async A copies.
// ---------------------------------------------------------------------------
__global__ __launch_bounds__(256)
void gemm_wmma(const _Float16* __restrict__ A, const _Float16* __restrict__ Bm,
               const float* __restrict__ bias, void* __restrict__ Cv,
               int M, int N, int K, int lda, int ldb, int ldc,
               float scale, int mode)
{
    __shared__ __align__(16) _Float16 ldsA[2][128 * LROW];   // row-major [m][k]
    __shared__ __align__(16) _Float16 ldsB[2][128 * LROW];   // col-major [n][k]

    const int tid  = threadIdx.x;
    const int lane = tid & 31;
    const int wid  = tid >> 5;           // 0..7
    const int wm   = wid >> 1;           // 0..3 : 32-row group
    const int wn   = wid & 1;            // 0..1 : 64-col group

    const int tileM = blockIdx.y * 128;
    const int tileN = blockIdx.x * 128;

    // global->LDS mapping
    const int arow = tid >> 1;           // 0..127
    const int acol = (tid & 1) * 16;     // 0 or 16
    const int bk4  = (tid >> 5) * 4;     // k quad: 0..28
    const int bn4  = (tid & 31) * 4;     // n quad: 0..124

    const int mlo = lane & 15;
    const int h   = lane >> 4;

    v8f zero = {};
    v8f acc[2][4];
    #pragma unroll
    for (int i = 0; i < 2; ++i)
        #pragma unroll
        for (int j = 0; j < 4; ++j) acc[i][j] = zero;

    const int nk = K / 32;
    const _Float16* gA = A + (size_t)(tileM + arow) * lda + acol;

    uint4 ra0, ra1;     // A staging (fallback path)
    uint2 rb[4];        // B staging (4 rows x 4 cols)

    auto issueLoads = [&](int k0, int buf) {
#if USE_ASYNC_LDS
        const _Float16* g = gA + k0;
        _Float16* l = &ldsA[buf][arow * LROW + acol];
        __builtin_amdgcn_global_load_async_to_lds_b128((as1_v4p)g, (as3_v4p)l, 0, 0);
        __builtin_amdgcn_global_load_async_to_lds_b128((as1_v4p)(g + 8), (as3_v4p)(l + 8), 0, 0);
#else
        const uint4* ga = (const uint4*)(gA + k0);
        ra0 = ga[0]; ra1 = ga[1];
#endif
        #pragma unroll
        for (int r = 0; r < 4; ++r)
            rb[r] = *(const uint2*)(Bm + (size_t)(k0 + bk4 + r) * ldb + tileN + bn4);
    };

    auto commit = [&](int buf) {
#if USE_ASYNC_LDS
        __builtin_amdgcn_s_wait_asynccnt(0);
#else
        *(uint4*)&ldsA[buf][arow * LROW + acol]     = ra0;
        *(uint4*)&ldsA[buf][arow * LROW + acol + 8] = ra1;
#endif
        // transpose B 4x4 block in registers -> K-packed ds_store_b64
        union { uint2 u; _Float16 hh[4]; } rr[4];
        #pragma unroll
        for (int r = 0; r < 4; ++r) rr[r].u = rb[r];
        #pragma unroll
        for (int c = 0; c < 4; ++c) {
            union { _Float16 hh[4]; uint2 u; } pk;
            #pragma unroll
            for (int r = 0; r < 4; ++r) pk.hh[r] = rr[r].hh[c];
            *(uint2*)&ldsB[buf][(bn4 + c) * LROW + bk4] = pk.u;
        }
    };

    // prologue
    issueLoads(0, 0);
    commit(0);
    __syncthreads();

    for (int kt = 0; kt < nk; ++kt) {
        const int buf = kt & 1;
        if (kt + 1 < nk) issueLoads((kt + 1) * 32, buf ^ 1);   // overlap with WMMAs

        v16h bf[4];
        #pragma unroll
        for (int j = 0; j < 4; ++j) {
            const uint4* cp = (const uint4*)&ldsB[buf][(wn * 64 + j * 16 + mlo) * LROW];
            bf[j] = frag16(cp[h], cp[2 + h]);
        }
        #pragma unroll
        for (int i = 0; i < 2; ++i) {
            const uint4* rp = (const uint4*)&ldsA[buf][(wm * 32 + i * 16 + mlo) * LROW];
            v16h af = frag16(rp[h], rp[2 + h]);
            #pragma unroll
            for (int j = 0; j < 4; ++j)
                acc[i][j] = wmma16(af, bf[j], acc[i][j]);
        }

        if (kt + 1 < nk) commit(buf ^ 1);
        __syncthreads();
    }

    // ---- store: lane owns column n (= mlo part), rows m = v + 8*h
    #pragma unroll
    for (int i = 0; i < 2; ++i) {
        const int mbase = tileM + wm * 32 + i * 16 + h * 8;
        #pragma unroll
        for (int j = 0; j < 4; ++j) {
            const int ng = tileN + wn * 64 + j * 16 + mlo;
            const float bv = bias ? bias[ng] : 0.0f;
            if (mode == 0) {
                float* Cf = (float*)Cv;
                #pragma unroll
                for (int v = 0; v < 8; ++v)
                    Cf[(size_t)(mbase + v) * ldc + ng] = acc[i][j][v] * scale + bv;
            } else if (mode == 1) {
                _Float16* Ch = (_Float16*)Cv;
                #pragma unroll
                for (int v = 0; v < 8; ++v)
                    Ch[(size_t)(mbase + v) * ldc + ng] =
                        (_Float16)(acc[i][j][v] * scale + bv);
            } else { // mode 2: batched transposed f16 store (K^T)
                const int bidx = mbase / ldc;            // block never straddles batches
                const int ms   = mbase - bidx * ldc;
                _Float16* Ct = (_Float16*)Cv + ((size_t)bidx * N + ng) * ldc + ms;
                union { _Float16 hh[8]; uint4 u; } pk;
                #pragma unroll
                for (int v = 0; v < 8; ++v)
                    pk.hh[v] = (_Float16)(acc[i][j][v] * scale + bv);
                *(uint4*)Ct = pk.u;                      // 8 consecutive m -> one b128
            }
        }
    }
}

// ---------------------------------------------------------------------------
// row-wise softmax: f32 [rows, ncols] -> f16, one block per row
// ---------------------------------------------------------------------------
__global__ __launch_bounds__(256)
void softmax_rows(const float* __restrict__ S, _Float16* __restrict__ P, int ncols)
{
    __shared__ float buf[256];
    const int tid = threadIdx.x;
    const float* sr = S + (size_t)blockIdx.x * ncols;
    _Float16*    pr = P + (size_t)blockIdx.x * ncols;

    float lmax = -3.0e38f;
    for (int c = tid; c < ncols; c += 256) lmax = fmaxf(lmax, sr[c]);
    buf[tid] = lmax; __syncthreads();
    for (int s = 128; s > 0; s >>= 1) {
        if (tid < s) buf[tid] = fmaxf(buf[tid], buf[tid + s]);
        __syncthreads();
    }
    const float rmax = buf[0];
    __syncthreads();

    float lsum = 0.0f;
    for (int c = tid; c < ncols; c += 256) lsum += __expf(sr[c] - rmax);
    buf[tid] = lsum; __syncthreads();
    for (int s = 128; s > 0; s >>= 1) {
        if (tid < s) buf[tid] += buf[tid + s];
        __syncthreads();
    }
    const float inv = 1.0f / buf[0];
    __syncthreads();

    for (int c = tid; c < ncols; c += 256)
        pr[c] = (_Float16)(__expf(sr[c] - rmax) * inv);
}

// ---------------------------------------------------------------------------
// host orchestration
// ---------------------------------------------------------------------------
static inline void launch_gemm(const _Float16* A, const _Float16* Bm, const float* bias,
                               void* C, int M, int N, int K, int lda, int ldb, int ldc,
                               float scale, int mode, hipStream_t s)
{
    dim3 grid(N / 128, M / 128);
    gemm_wmma<<<grid, 256, 0, s>>>(A, Bm, bias, C, M, N, K, lda, ldb, ldc, scale, mode);
}

extern "C" void kernel_launch(void* const* d_in, const int* in_sizes, int n_in,
                              void* d_out, int out_size, void* d_ws, size_t ws_size,
                              hipStream_t stream)
{
    const float* x  = (const float*)d_in[0];
    const float* Wq = (const float*)d_in[1];
    const float* bq = (const float*)d_in[2];
    const float* Wk = (const float*)d_in[3];
    const float* bk = (const float*)d_in[4];
    const float* Wv = (const float*)d_in[5];
    const float* bv = (const float*)d_in[6];
    const float* Wp = (const float*)d_in[7];
    const float* bp = (const float*)d_in[8];

    const int MSD = NB * SS;                   // 16384 rows
    const size_t nX   = (size_t)MSD * DD;      // 12,582,912
    const size_t nW   = (size_t)DD * DD;       // 589,824
    const size_t nSS  = (size_t)SS * SS;       // 4,194,304

    // workspace layout (halfs first, then f32 scores scratch)
    _Float16* xf16 = (_Float16*)d_ws;
    _Float16* Wqt  = xf16 + nX;
    _Float16* Wkt  = Wqt + nW;
    _Float16* Wvt  = Wkt + nW;
    _Float16* Wpt  = Wvt + nW;
    _Float16* Qf   = Wpt + nW;                 // [B*S, D]
    _Float16* Ktf  = Qf  + nX;                 // [B, D, S]
    _Float16* Vf   = Ktf + nX;                 // [B*S, D]
    _Float16* Attn = Vf  + nX;                 // [B*S, D]
    _Float16* Pf   = Attn + nX;                // [S, S] per-batch scratch
    float*    Sc   = (float*)(Pf + nSS);       // [S, S] f32 per-batch scratch

    const float invscale = 1.0f / sqrtf((float)DD);

    // 1) x -> f16
    cvt_f32_to_f16<<<(int)(nX / 4 / 256), 256, 0, stream>>>(x, xf16, (int)nX);

    // 2) weight transposes -> f16 (Wt[d][e] = W[e][d])
    {
        dim3 g(DD / 32, DD / 32);
        transpose_to_f16<<<g, 256, 0, stream>>>(Wq, Wqt, DD);
        transpose_to_f16<<<g, 256, 0, stream>>>(Wk, Wkt, DD);
        transpose_to_f16<<<g, 256, 0, stream>>>(Wv, Wvt, DD);
        transpose_to_f16<<<g, 256, 0, stream>>>(Wp, Wpt, DD);
    }

    // 3) projections (M = 16384, N = K = 768)
    launch_gemm(xf16, Wqt, bq, Qf,  MSD, DD, DD, DD, DD, DD, 1.0f, 1, stream); // Q f16
    launch_gemm(xf16, Wkt, bk, Ktf, MSD, DD, DD, DD, DD, SS, 1.0f, 2, stream); // K^T f16 [b][d][s]
    launch_gemm(xf16, Wvt, bv, Vf,  MSD, DD, DD, DD, DD, DD, 1.0f, 1, stream); // V f16

    // 4) attention per batch (score scratch stays L2-resident)
    for (int b = 0; b < NB; ++b) {
        const _Float16* Qb  = Qf  + (size_t)b * SS * DD;
        const _Float16* Ktb = Ktf + (size_t)b * DD * SS;
        const _Float16* Vb  = Vf  + (size_t)b * SS * DD;
        _Float16*       Ab  = Attn + (size_t)b * SS * DD;

        // scores = (Q @ K^T) / sqrt(D)  : [S,S] f32
        launch_gemm(Qb, Ktb, nullptr, Sc, SS, SS, DD, DD, SS, SS, invscale, 0, stream);
        // softmax rows -> f16 P
        softmax_rows<<<SS, 256, 0, stream>>>(Sc, Pf, SS);
        // attn = P @ V : [S, D] f16
        launch_gemm(Pf, Vb, nullptr, Ab, SS, DD, SS, SS, DD, DD, 1.0f, 1, stream);
    }

    // 5) output projection: out = Attn @ Wp^T + bp : f32
    launch_gemm(Attn, Wpt, bp, d_out, MSD, DD, DD, DD, DD, DD, 1.0f, 0, stream);
}
